// LoRAAttnProcessor_45732811767996
// MI455X (gfx1250) — matrix operand order repaired and checked by reading, not compiled, not run on hardware
//
#include <hip/hip_runtime.h>

// ---------------------------------------------------------------------------
// LoRA attention for MI455X (gfx1250, wave32, WMMA).
// LoRA is folded into effective weights: W_eff = W + up@down, so the whole op
// is 4 GEMMs + flash attention. All matmuls run on v_wmma_f32_16x16x32_bf16
// with a 3-term bf16 hi/lo split for ~fp32 precision, fp32 accumulate.
// Workspace requirement: ~236 MB (weights hi/lo + activation hi/lo planes).
// ---------------------------------------------------------------------------

#define DEV __device__ __forceinline__

typedef __attribute__((ext_vector_type(16))) __bf16 v16bf;
typedef __attribute__((ext_vector_type(8)))  __bf16 bf16x8;
typedef __attribute__((ext_vector_type(8)))  float  v8f;

constexpr int Bc  = 4;
constexpr int Sc  = 2048;
constexpr int Dc  = 1280;
constexpr int Hc  = 8;
constexpr int HDc = 160;
constexpr int BSc = Bc * Sc;                 // 8192 rows
constexpr float SCALE = 0.07905694150420949f; // 160^-0.5

DEV bf16x8 ld8(const __bf16* p) { return *(const bf16x8*)p; }
DEV v16bf mk16(bf16x8 a, bf16x8 b) {
  union { v16bf v; bf16x8 h[2]; } u; u.h[0] = a; u.h[1] = b; return u.v;
}
// A-fragment (16x32): lane<16 -> K {k0..k0+7, k0+16..k0+23}; lane>=16 -> +8
DEV v16bf ldA(const __bf16* row, int k0, int half) {
  return mk16(ld8(row + k0 + half * 8), ld8(row + k0 + 16 + half * 8));
}
// B-fragment (32x16): lane<16 -> K {k0..k0+7, k0+16..k0+23}; lane>=16 -> +8
DEV v16bf ldB(const __bf16* row, int k0, int half) {
  return mk16(ld8(row + k0 + half * 8), ld8(row + k0 + 16 + half * 8));
}
DEV void split(float x, __bf16& h, __bf16& l) {
  h = (__bf16)x;
  l = (__bf16)(x - (float)h);
}
// 3-term split product: (ah+al)*(bh+bl) ~= ah*bh + ah*bl + al*bh
DEV v8f wmma3(v16bf ah, v16bf al, v16bf bh, v16bf bl, v8f c) {
  c = __builtin_amdgcn_wmma_f32_16x16x32_bf16(false, ah, false, bh, (short)0, c, false, false);
  c = __builtin_amdgcn_wmma_f32_16x16x32_bf16(false, ah, false, bl, (short)0, c, false, false);
  c = __builtin_amdgcn_wmma_f32_16x16x32_bf16(false, al, false, bh, (short)0, c, false, false);
  return c;
}

// ---------------------------------------------------------------------------
// Prep: W_eff = W + up@down  ->  bf16 hi/lo planes, row-major [n][k]
// ---------------------------------------------------------------------------
__global__ void build_weff(const float* __restrict__ W,
                           const float* __restrict__ up,    // [D,4]
                           const float* __restrict__ down,  // [4,D]
                           __bf16* __restrict__ oh, __bf16* __restrict__ ol) {
  const int idx = blockIdx.x * blockDim.x + threadIdx.x;
  if (idx >= Dc * Dc) return;
  const int n = idx / Dc, k = idx - n * Dc;
  float w = W[idx];
  #pragma unroll
  for (int r = 0; r < 4; ++r) w += up[n * 4 + r] * down[r * Dc + k];
  __bf16 h, l; split(w, h, l);
  oh[idx] = h; ol[idx] = l;
}

__global__ void cvt_planes(const float* __restrict__ x,
                           __bf16* __restrict__ hi, __bf16* __restrict__ lo, int n) {
  const int idx = blockIdx.x * blockDim.x + threadIdx.x;
  if (idx >= n) return;
  __bf16 h, l; split(x[idx], h, l);
  hi[idx] = h; lo[idx] = l;
}

// ---------------------------------------------------------------------------
// GEMM: Y[M=8192, N=1280] = X[M,K=1280] @ Weff^T, bf16-split WMMA.
// Block: 256 thr = 8 waves; tile 128(M) x 64(N); each wave one 16x64 strip.
// mode 0: write Y as bf16 hi/lo planes [M][D]            (Q, K projections)
// mode 1: write fp32 Y + bias to outF                    (output projection)
// mode 2: write transposed per-head planes [b,h][d][S]   (V projection)
// ---------------------------------------------------------------------------
__global__ __launch_bounds__(256) void gemm_split_bf16(
    const __bf16* __restrict__ Xhi, const __bf16* __restrict__ Xlo,
    const __bf16* __restrict__ Whi, const __bf16* __restrict__ Wlo,
    const float*  __restrict__ bias,
    float* __restrict__ outF,
    __bf16* __restrict__ outHi, __bf16* __restrict__ outLo,
    int mode) {
  const int tid = threadIdx.x;
  const int lane = tid & 31, wave = tid >> 5;
  const int laneN = lane & 15, half = lane >> 4;
  const int nBase = blockIdx.x * 64;
  const int mBase = blockIdx.y * 128;
  const int mRow  = mBase + wave * 16 + laneN;

  const __bf16* xh = Xhi + (size_t)mRow * Dc;
  const __bf16* xl = Xlo + (size_t)mRow * Dc;

  v8f acc[4] = {};
  #pragma unroll 1
  for (int k0 = 0; k0 < Dc; k0 += 32) {
    __builtin_prefetch(xh + k0 + 64, 0, 1);   // -> global_prefetch_b8
    v16bf ah = ldA(xh, k0, half);
    v16bf al = ldA(xl, k0, half);
    #pragma unroll
    for (int t = 0; t < 4; ++t) {
      const int n = nBase + t * 16 + laneN;
      const __bf16* wrh = Whi + (size_t)n * Dc;
      const __bf16* wrl = Wlo + (size_t)n * Dc;
      v16bf bh = ldB(wrh, k0, half);
      v16bf bl = ldB(wrl, k0, half);
      acc[t] = wmma3(ah, al, bh, bl, acc[t]);
    }
  }

  const int rBase = mBase + wave * 16 + 8 * half;  // D-frag rows: rBase + r
  if (mode == 0) {
    #pragma unroll
    for (int t = 0; t < 4; ++t) {
      const int col = nBase + t * 16 + laneN;
      #pragma unroll
      for (int r = 0; r < 8; ++r) {
        __bf16 h, l; split(acc[t][r], h, l);
        const size_t off = (size_t)(rBase + r) * Dc + col;
        outHi[off] = h; outLo[off] = l;
      }
    }
  } else if (mode == 1) {
    #pragma unroll
    for (int t = 0; t < 4; ++t) {
      const int col = nBase + t * 16 + laneN;
      const float bv = bias[col];
      #pragma unroll
      for (int r = 0; r < 8; ++r)
        outF[(size_t)(rBase + r) * Dc + col] = acc[t][r] + bv;
    }
  } else {
    // transposed V: Vt[(b*H+h)*HD + d][s], 8 consecutive s per lane -> b128
    const int b  = rBase >> 11;       // / 2048
    const int s0 = rBase & 2047;
    #pragma unroll
    for (int t = 0; t < 4; ++t) {
      const int n = nBase + t * 16 + laneN;
      const int hh = n / HDc, d = n - hh * HDc;
      const size_t base = ((size_t)(b * Hc + hh) * HDc + d) * Sc + s0;
      bf16x8 vh, vl;
      #pragma unroll
      for (int r = 0; r < 8; ++r) {
        __bf16 ph, pl; split(acc[t][r], ph, pl);
        vh[r] = ph; vl[r] = pl;
      }
      *(bf16x8*)(outHi + base) = vh;
      *(bf16x8*)(outLo + base) = vl;
    }
  }
}

// ---------------------------------------------------------------------------
// Flash attention. Block = 4 waves; wave owns a 16-row query block of one
// (b,h). 64 iterations over 32-key tiles: QK^T (WMMA) -> online softmax
// (shfl_xor row reductions) -> P restaged D->A layout through LDS -> P@V
// (WMMA against transposed V planes). Writes ctx as bf16 hi/lo planes.
// ---------------------------------------------------------------------------
__global__ __launch_bounds__(128) void flash_attn(
    const __bf16* __restrict__ Qhi, const __bf16* __restrict__ Qlo,
    const __bf16* __restrict__ Khi, const __bf16* __restrict__ Klo,
    const __bf16* __restrict__ Vthi, const __bf16* __restrict__ Vtlo,
    __bf16* __restrict__ Chi, __bf16* __restrict__ Clo) {
  __shared__ __align__(16) __bf16 sPh[4][16][32];
  __shared__ __align__(16) __bf16 sPl[4][16][32];

  const int tid = threadIdx.x;
  const int lane = tid & 31, wave = tid >> 5;
  const int laneN = lane & 15, half = lane >> 4;
  const int bh = blockIdx.y, b = bh >> 3, h = bh & 7;
  const int sq = blockIdx.x * 64 + wave * 16;

  const __bf16* qrh = Qhi + ((size_t)(b * Sc + sq + laneN) * Dc + h * HDc);
  const __bf16* qrl = Qlo + ((size_t)(b * Sc + sq + laneN) * Dc + h * HDc);

  v8f ctx[10] = {};
  float mrow[8], lrow[8];
  #pragma unroll
  for (int r = 0; r < 8; ++r) { mrow[r] = -3.0e38f; lrow[r] = 0.f; }

  #pragma unroll 1
  for (int kb = 0; kb < Sc; kb += 32) {
    const size_t kr0 = (size_t)(b * Sc + kb + laneN) * Dc + h * HDc;
    const size_t kr1 = (size_t)(b * Sc + kb + 16 + laneN) * Dc + h * HDc;
    v8f c0 = {}, c1 = {};
    #pragma unroll
    for (int ks = 0; ks < 5; ++ks) {
      const int d0 = ks * 32;
      v16bf qh = ldA(qrh, d0, half);
      v16bf ql = ldA(qrl, d0, half);
      v16bf b0h = ldB(Khi + kr0, d0, half);
      v16bf b0l = ldB(Klo + kr0, d0, half);
      c0 = wmma3(qh, ql, b0h, b0l, c0);
      v16bf b1h = ldB(Khi + kr1, d0, half);
      v16bf b1l = ldB(Klo + kr1, d0, half);
      c1 = wmma3(qh, ql, b1h, b1l, c1);
    }

    // Online softmax. D-frag row r belongs to query row sq + 8*half + r;
    // its 16 columns live across the 16 lanes of this half -> shfl_xor.
    float alf[8];
    #pragma unroll
    for (int r = 0; r < 8; ++r) {
      const float s0 = c0[r] * SCALE;
      const float s1 = c1[r] * SCALE;
      float tm = fmaxf(s0, s1);
      #pragma unroll
      for (int m = 1; m < 16; m <<= 1) tm = fmaxf(tm, __shfl_xor(tm, m, 32));
      const float mn = fmaxf(mrow[r], tm);
      const float alpha = __expf(mrow[r] - mn);
      const float p0 = __expf(s0 - mn);
      const float p1 = __expf(s1 - mn);
      float rs = p0 + p1;
      #pragma unroll
      for (int m = 1; m < 16; m <<= 1) rs += __shfl_xor(rs, m, 32);
      lrow[r] = lrow[r] * alpha + rs;
      mrow[r] = mn;
      alf[r] = alpha;
      __bf16 ph, pl;
      split(p0, ph, pl);
      sPh[wave][8 * half + r][laneN] = ph;
      sPl[wave][8 * half + r][laneN] = pl;
      split(p1, ph, pl);
      sPh[wave][8 * half + r][16 + laneN] = ph;
      sPl[wave][8 * half + r][16 + laneN] = pl;
    }
    #pragma unroll
    for (int t = 0; t < 10; ++t) {
      #pragma unroll
      for (int r = 0; r < 8; ++r) ctx[t][r] *= alf[r];
    }

    // All DS stores above are from this wave only; wait then read A-layout.
    asm volatile("s_wait_dscnt 0" ::: "memory");
    v16bf pah = mk16(*(const bf16x8*)&sPh[wave][laneN][half * 8],
                     *(const bf16x8*)&sPh[wave][laneN][16 + half * 8]);
    v16bf pal = mk16(*(const bf16x8*)&sPl[wave][laneN][half * 8],
                     *(const bf16x8*)&sPl[wave][laneN][16 + half * 8]);

    #pragma unroll
    for (int t = 0; t < 10; ++t) {
      const int d = t * 16 + laneN;
      const __bf16* vrh = Vthi + ((size_t)(bh * HDc + d) * Sc + kb);
      const __bf16* vrl = Vtlo + ((size_t)(bh * HDc + d) * Sc + kb);
      v16bf vbh = ldB(vrh, 0, half);
      v16bf vbl = ldB(vrl, 0, half);
      ctx[t] = wmma3(pah, pal, vbh, vbl, ctx[t]);
    }
  }

  // Normalize and emit ctx as bf16 hi/lo planes in [b,s,D] layout.
  #pragma unroll
  for (int t = 0; t < 10; ++t) {
    const int d = t * 16 + laneN;
    #pragma unroll
    for (int r = 0; r < 8; ++r) {
      const float y = ctx[t][r] / lrow[r];
      const int s = sq + 8 * half + r;
      const size_t off = (size_t)(b * Sc + s) * Dc + h * HDc + d;
      __bf16 ph, pl; split(y, ph, pl);
      Chi[off] = ph; Clo[off] = pl;
    }
  }
}

// ---------------------------------------------------------------------------
extern "C" void kernel_launch(void* const* d_in, const int* in_sizes, int n_in,
                              void* d_out, int out_size, void* d_ws, size_t ws_size,
                              hipStream_t stream) {
  (void)in_sizes; (void)n_in; (void)out_size; (void)ws_size;
  const float* x      = (const float*)d_in[0];
  const float* Wq     = (const float*)d_in[1];
  const float* Wk     = (const float*)d_in[2];
  const float* Wv     = (const float*)d_in[3];
  const float* Wo     = (const float*)d_in[4];
  const float* bo     = (const float*)d_in[5];
  const float* q_down = (const float*)d_in[6];
  const float* q_up   = (const float*)d_in[7];
  const float* k_down = (const float*)d_in[8];
  const float* k_up   = (const float*)d_in[9];
  const float* v_down = (const float*)d_in[10];
  const float* v_up   = (const float*)d_in[11];
  const float* o_down = (const float*)d_in[12];
  const float* o_up   = (const float*)d_in[13];

  const size_t WW = (size_t)Dc * Dc;   // 1.64M elems
  const size_t XX = (size_t)BSc * Dc;  // 10.5M elems
  __bf16* p = (__bf16*)d_ws;
  __bf16 *WqH = p, *WqL = p + WW, *WkH = p + 2*WW, *WkL = p + 3*WW,
         *WvH = p + 4*WW, *WvL = p + 5*WW, *WoH = p + 6*WW, *WoL = p + 7*WW;
  p += 8 * WW;
  __bf16 *XH = p, *XL = p + XX;           p += 2 * XX;
  __bf16 *QH = p, *QL = p + XX;           p += 2 * XX;
  __bf16 *KH = p, *KL = p + XX;           p += 2 * XX;
  __bf16 *VtH = p, *VtL = p + XX;         p += 2 * XX;
  __bf16 *CH = p, *CL = p + XX;           p += 2 * XX;

  const int wBlocks = (int)((WW + 255) / 256);
  build_weff<<<wBlocks, 256, 0, stream>>>(Wq, q_up, q_down, WqH, WqL);
  build_weff<<<wBlocks, 256, 0, stream>>>(Wk, k_up, k_down, WkH, WkL);
  build_weff<<<wBlocks, 256, 0, stream>>>(Wv, v_up, v_down, WvH, WvL);
  build_weff<<<wBlocks, 256, 0, stream>>>(Wo, o_up, o_down, WoH, WoL);
  cvt_planes<<<(int)((XX + 255) / 256), 256, 0, stream>>>(x, XH, XL, (int)XX);

  dim3 gg(Dc / 64, BSc / 128);
  gemm_split_bf16<<<gg, 256, 0, stream>>>(XH, XL, WqH, WqL, nullptr, nullptr, QH, QL, 0);
  gemm_split_bf16<<<gg, 256, 0, stream>>>(XH, XL, WkH, WkL, nullptr, nullptr, KH, KL, 0);
  gemm_split_bf16<<<gg, 256, 0, stream>>>(XH, XL, WvH, WvL, nullptr, nullptr, VtH, VtL, 2);

  flash_attn<<<dim3(Sc / 64, Bc * Hc), 128, 0, stream>>>(QH, QL, KH, KL, VtH, VtL, CH, CL);

  gemm_split_bf16<<<gg, 256, 0, stream>>>(CH, CL, WoH, WoL, bo, (float*)d_out,
                                          nullptr, nullptr, 1);
}
